// TRecOnlyFBP_16527034155484
// MI455X (gfx1250) — compile-verified, matrix operand order checked by reading, not run on hardware
//
#include <hip/hip_runtime.h>
#include <math.h>

// ---------------- problem constants ----------------
#define BB   4
#define SS   8320
#define DD   512
#define NHH  8
#define DHH  64
#define LL   4
#define FFN  2048
#define IMGN 128
#define WHN  65
#define DCN  8
#define MM   (BB*SS)          // 33280 rows

typedef __attribute__((ext_vector_type(16))) _Float16 v16h;
typedef __attribute__((ext_vector_type(8)))  _Float16 v8h;
typedef __attribute__((ext_vector_type(8)))  float    v8f;

// =====================================================================
// Async global->LDS copy (CDNA5): per-lane 16B, tracked by ASYNCcnt.
// =====================================================================
__device__ __forceinline__ void async_load_b128(const void* gptr, unsigned int lds_off) {
  asm volatile("global_load_async_to_lds_b128 %0, %1, off"
               :: "v"(lds_off), "v"(gptr)
               : "memory");
}

// =====================================================================
// WMMA GEMM:  C[M,N] = A[M,K] (f16) x Wt[N,K] (f16, pre-transposed) + bias.
// Block tile 128x128, BK=32, 256 threads = 8 waves (2x4), wave tile 64x32.
// Double-buffered LDS filled via global_load_async_to_lds_b128; last
// iteration peeled so the steady-state loop is branch-free.
// =====================================================================
#define GBM 128
#define GBN 128
#define GBK 32
#define LDT 40            // LDS row stride in halves (80B rows, 16B-aligned)
#define TILEB (GBM*LDT*2) // 10240 bytes per tile buffer

__global__ __launch_bounds__(256) void gemm_wmma_kernel(
    const _Float16* __restrict__ A,   // M x K
    const _Float16* __restrict__ Wt,  // N x K (transposed weights)
    const float* __restrict__ bias,
    float* __restrict__ C,            // nullable
    _Float16* __restrict__ Ch,        // nullable
    int M, int N, int K, int relu)
{
  __shared__ __align__(16) _Float16 As[2][GBM * LDT];
  __shared__ __align__(16) _Float16 Bs[2][GBN * LDT];

  const int tid     = threadIdx.x;
  const int lane    = tid & 31;
  const int wave    = tid >> 5;
  const int wm      = wave >> 2;          // 0..1  (M dir)
  const int wn      = wave & 3;           // 0..3  (N dir)
  const int laneRow = lane & 15;
  const int laneHalf= lane >> 4;          // 0/1
  const int kb      = laneHalf * 8;       // k base within BK per ISA frag layout
  const int mBase   = blockIdx.y * GBM;
  const int nBase   = blockIdx.x * GBN;

  // async-fill assignment: 512 16B chunks per 8KB tile -> 2 A + 2 B per thread
  const int ch0   = tid * 2;
  const int rowA0 = ch0 >> 2,       ccA0 = ch0 & 3;
  const int rowA1 = (ch0 + 1) >> 2, ccA1 = (ch0 + 1) & 3;

  // per-lane source pointers (advance by 64B = GBK halves per stage)
  const char* gA0 = (const char*)(A  + (size_t)(mBase + rowA0) * K) + ccA0 * 16;
  const char* gA1 = (const char*)(A  + (size_t)(mBase + rowA1) * K) + ccA1 * 16;
  const char* gB0 = (const char*)(Wt + (size_t)(nBase + rowA0) * K) + ccA0 * 16;
  const char* gB1 = (const char*)(Wt + (size_t)(nBase + rowA1) * K) + ccA1 * 16;
  // per-lane LDS destinations (buffer 0); buffer 1 = +TILEB bytes
  const unsigned la0 = (unsigned)(size_t)&As[0][rowA0 * LDT + ccA0 * 8];
  const unsigned la1 = (unsigned)(size_t)&As[0][rowA1 * LDT + ccA1 * 8];
  const unsigned lb0 = (unsigned)(size_t)&Bs[0][rowA0 * LDT + ccA0 * 8];
  const unsigned lb1 = (unsigned)(size_t)&Bs[0][rowA1 * LDT + ccA1 * 8];

  v8f acc[4][2] = {};
  const int nt = K / GBK;                 // >= 2 for all our shapes

  auto issue = [&](int buf, int stage) {
    const int      go = stage * (GBK * 2);      // 64B per stage
    const unsigned lo = buf * TILEB;
    async_load_b128(gA0 + go, la0 + lo);
    async_load_b128(gB0 + go, lb0 + lo);
    async_load_b128(gA1 + go, la1 + lo);
    async_load_b128(gB1 + go, lb1 + lo);
  };

  auto compute = [&](int buf) {
    v16h afrag[4], bfrag[2];
    #pragma unroll
    for (int mi = 0; mi < 4; ++mi) {
      int r = wm * 64 + mi * 16 + laneRow;
      v8h lo = *(const v8h*)&As[buf][r * LDT + kb];
      v8h hi = *(const v8h*)&As[buf][r * LDT + kb + 16];
      afrag[mi] = __builtin_shufflevector(lo, hi, 0,1,2,3,4,5,6,7,8,9,10,11,12,13,14,15);
    }
    #pragma unroll
    for (int ni = 0; ni < 2; ++ni) {
      int cidx = wn * 32 + ni * 16 + laneRow;
      v8h lo = *(const v8h*)&Bs[buf][cidx * LDT + kb];
      v8h hi = *(const v8h*)&Bs[buf][cidx * LDT + kb + 16];
      bfrag[ni] = __builtin_shufflevector(lo, hi, 0,1,2,3,4,5,6,7,8,9,10,11,12,13,14,15);
    }
    #pragma unroll
    for (int mi = 0; mi < 4; ++mi)
      #pragma unroll
      for (int ni = 0; ni < 2; ++ni)
        acc[mi][ni] = __builtin_amdgcn_wmma_f32_16x16x32_f16(
            false, afrag[mi], false, bfrag[ni], (short)0, acc[mi][ni], false, false);
  };

  issue(0, 0);
  for (int t = 0; t < nt - 1; ++t) {
    issue((t + 1) & 1, t + 1);
    // async loads complete in order: <=4 outstanding => current buffer ready
    asm volatile("s_wait_asynccnt 0x4" ::: "memory");
    __syncthreads();
    compute(t & 1);
    __syncthreads();            // all waves done reading buf before refill
  }
  asm volatile("s_wait_asynccnt 0x0" ::: "memory");
  __syncthreads();
  compute((nt - 1) & 1);

  // ---- epilogue: D layout => lane<16: col=lane, rows 0..7; lane>=16: rows 8..15
  #pragma unroll
  for (int mi = 0; mi < 4; ++mi) {
    #pragma unroll
    for (int ni = 0; ni < 2; ++ni) {
      int col = nBase + wn * 32 + ni * 16 + laneRow;
      float bv = bias[col];
      #pragma unroll
      for (int j = 0; j < 8; ++j) {
        int row = mBase + wm * 64 + mi * 16 + laneHalf * 8 + j;
        float v = acc[mi][ni][j] + bv;
        if (relu) v = fmaxf(v, 0.f);
        if (C)  C[(size_t)row * N + col] = v;
        if (Ch) Ch[(size_t)row * N + col] = (_Float16)v;
      }
    }
  }
}

// =====================================================================
// Weight convert + transpose: W[K,N] f32 -> Wt[N,K] f16
// =====================================================================
__global__ void convT_kernel(const float* __restrict__ W, _Float16* __restrict__ Wt,
                             int K, int N)
{
  int idx = blockIdx.x * 256 + threadIdx.x;
  if (idx >= K * N) return;
  int n = idx / K, k = idx - n * K;
  Wt[idx] = (_Float16)W[(size_t)k * N + n];
}

// =====================================================================
// Embedding + 2D sinusoidal positional encoding (f32 + f16 copies)
// =====================================================================
__global__ void embed_kernel(const float* __restrict__ fbp, const int* __restrict__ coords,
                             const float* __restrict__ emb_w, const float* __restrict__ emb_b,
                             float* __restrict__ H, _Float16* __restrict__ Hh)
{
  int idx = blockIdx.x * 256 + threadIdx.x;
  if (idx >= MM * DD) return;
  int c  = idx & (DD - 1);
  int bs = idx >> 9;
  int s  = bs % SS;
  float out;
  if (c < 256) {
    float f0 = fbp[(size_t)bs * 2], f1 = fbp[(size_t)bs * 2 + 1];
    out = f0 * emb_w[c] + f1 * emb_w[256 + c] + emb_b[c];
  } else {
    int cc = c - 256;
    int p  = (cc < 128) ? coords[s * 2 + 1] : coords[s * 2];  // col half then row half
    int j  = cc & 127;
    int i  = j >> 1;
    float div = expf(-(float)(2 * i) * (logf(10000.f) / 128.f));
    float a = (float)p * div;
    out = (j & 1) ? cosf(a) : sinf(a);
  }
  H[idx] = out;
  Hh[idx] = (_Float16)out;
}

// =====================================================================
// Linear attention
// =====================================================================
__device__ __forceinline__ float phi_elu(float x) { return x > 0.f ? x + 1.f : expf(x); }

__global__ __launch_bounds__(256) void kv_kernel(const float* __restrict__ Kb,
                                                 const float* __restrict__ Vb,
                                                 float* __restrict__ KV,
                                                 float* __restrict__ KSUM)
{
  __shared__ float lk[32][64];
  __shared__ float lv[32][64];
  int bh = blockIdx.x;            // b*8 + h
  int b = bh >> 3, h = bh & 7;
  int t = threadIdx.x;
  int d  = t & 63;
  int mg = t >> 6;                // 0..3 -> 16 m's each
  float acc[16] = {};
  float ks = 0.f;
  for (int s0 = 0; s0 < SS; s0 += 32) {
    #pragma unroll
    for (int j = 0; j < 8; ++j) {
      int idx = t * 8 + j;        // 0..2047
      int ss = idx >> 6, dd = idx & 63;
      size_t g = ((size_t)(b * SS + s0 + ss)) * DD + h * 64 + dd;
      float kvv = Kb[g];
      lk[ss][dd] = phi_elu(kvv);
      lv[ss][dd] = Vb[g];
    }
    __syncthreads();
    #pragma unroll 4
    for (int ss = 0; ss < 32; ++ss) {
      float kf = lk[ss][d];
      if (mg == 0) ks += kf;
      const float* vr = &lv[ss][mg * 16];
      #pragma unroll
      for (int j = 0; j < 16; ++j) acc[j] += kf * vr[j];
    }
    __syncthreads();
  }
  float* kvout = KV + ((size_t)bh * 64 + d) * 64 + mg * 16;
  #pragma unroll
  for (int j = 0; j < 16; ++j) kvout[j] = acc[j];
  if (mg == 0) KSUM[bh * 64 + d] = ks;
}

// att[b,s,h,m] = (phi(q) . kv[:,m]) / (phi(q) . ksum + 1e-6)  -> f16 (GEMM A input)
__global__ __launch_bounds__(256) void att_kernel(const float* __restrict__ Q,
                                                  const float* __restrict__ KV,
                                                  const float* __restrict__ KSUM,
                                                  _Float16* __restrict__ Atth)
{
  __shared__ float qf[DD];
  __shared__ float zsh[NHH];
  int bs = blockIdx.x;
  int b  = bs / SS;
  int t  = threadIdx.x;
  for (int c = t; c < DD; c += 256) qf[c] = phi_elu(Q[(size_t)bs * DD + c]);
  __syncthreads();
  if (t < NHH) {
    const float* kx = KSUM + (b * NHH + t) * 64;
    float dsum = 0.f;
    #pragma unroll 8
    for (int d2 = 0; d2 < 64; ++d2) dsum += qf[t * 64 + d2] * kx[d2];
    zsh[t] = 1.f / (dsum + 1e-6f);
  }
  __syncthreads();
  for (int c = t; c < DD; c += 256) {
    int h = c >> 6, m = c & 63;
    const float* kvp = KV + ((size_t)(b * NHH + h) * 64) * 64 + m;
    float a = 0.f;
    #pragma unroll 8
    for (int d2 = 0; d2 < 64; ++d2) a += qf[h * 64 + d2] * kvp[(size_t)d2 * 64];
    Atth[(size_t)bs * DD + c] = (_Float16)(a * zsh[h]);
  }
}

// =====================================================================
// LayerNorm over D=512 with fused residual: out = LN(x + dlt), f32 (+opt f16)
// =====================================================================
__global__ __launch_bounds__(256) void ln_kernel(const float* __restrict__ X,
                                                 const float* __restrict__ Dlt,
                                                 const float* __restrict__ g,
                                                 const float* __restrict__ b,
                                                 float* __restrict__ Out,
                                                 _Float16* __restrict__ Outh,
                                                 int hasDelta)
{
  __shared__ float r1[256], r2[256];
  __shared__ float smean, srstd;
  int row = blockIdx.x;
  int t = threadIdx.x;
  const float* x = X + (size_t)row * DD;
  float v0 = x[t],       v1 = x[t + 256];
  if (hasDelta) { const float* dl = Dlt + (size_t)row * DD; v0 += dl[t]; v1 += dl[t + 256]; }
  r1[t] = v0 + v1;
  r2[t] = v0 * v0 + v1 * v1;
  __syncthreads();
  for (int off = 128; off > 0; off >>= 1) {
    if (t < off) { r1[t] += r1[t + off]; r2[t] += r2[t + off]; }
    __syncthreads();
  }
  if (t == 0) {
    float m = r1[0] * (1.f / DD);
    float var = r2[0] * (1.f / DD) - m * m;
    smean = m; srstd = rsqrtf(var + 1e-5f);
  }
  __syncthreads();
  float m = smean, rs = srstd;
  float o0 = (v0 - m) * rs * g[t]       + b[t];
  float o1 = (v1 - m) * rs * g[t + 256] + b[t + 256];
  float* o = Out + (size_t)row * DD;
  o[t] = o0; o[t + 256] = o1;
  if (Outh) {
    _Float16* oh = Outh + (size_t)row * DD;
    oh[t] = (_Float16)o0; oh[t + 256] = (_Float16)o1;
  }
}

// =====================================================================
// Heads + convert2DFT scatter
// =====================================================================
__global__ void zero_kernel(float* __restrict__ p, int n)
{
  int i = blockIdx.x * 256 + threadIdx.x;
  if (i < n) p[i] = 0.f;
}

__global__ __launch_bounds__(256) void head_kernel(
    const float* __restrict__ Hf, const float* __restrict__ ampw, const float* __restrict__ ampb,
    const float* __restrict__ phw, const float* __restrict__ phb,
    const float* __restrict__ mag_min, const float* __restrict__ mag_max,
    const int* __restrict__ dst, float* __restrict__ yhat,
    float* __restrict__ dre, float* __restrict__ dim_)
{
  __shared__ float ra[256], rp[256];
  int bs = blockIdx.x;
  int t = threadIdx.x;
  const float* hrow = Hf + (size_t)bs * DD;
  ra[t] = hrow[t] * ampw[t] + hrow[t + 256] * ampw[t + 256];
  rp[t] = hrow[t] * phw[t]  + hrow[t + 256] * phw[t + 256];
  __syncthreads();
  for (int off = 128; off > 0; off >>= 1) {
    if (t < off) { ra[t] += ra[t + off]; rp[t] += rp[t + off]; }
    __syncthreads();
  }
  if (t == 0) {
    float ya = ra[0] + ampb[0];
    float yp = tanhf(rp[0] + phb[0]);
    yhat[(size_t)bs * 2]     = ya;
    yhat[(size_t)bs * 2 + 1] = yp;
    float mmin = mag_min[0], mmax = mag_max[0];
    float amp = expf((ya + 1.f) * 0.5f * (mmax - mmin) + mmin);
    float ph  = yp * 3.14159265358979323846f;
    int s = bs % SS, b = bs / SS;
    int pos = dst[s];
    dre[(size_t)b * SS + pos] = amp * cosf(ph);
    dim_[(size_t)b * SS + pos] = amp * sinf(ph);
  }
}

// =====================================================================
// irfftn(s=(128,128)): pass1 inverse DFT along axis1 (attenuation fused),
// pass2 irfft along axis2 (65 -> 128 real) with roll fused into the store.
// =====================================================================
__global__ __launch_bounds__(128) void ifft_col_kernel(
    const float* __restrict__ dre, const float* __restrict__ dim_,
    const float* __restrict__ att, float* __restrict__ yre, float* __restrict__ yim)
{
  __shared__ float xr[IMGN], xi[IMGN];
  int bc = blockIdx.x;               // b*65 + c
  int b = bc / WHN, c = bc % WHN;
  int t = threadIdx.x;
  float a = att[t * WHN + c];
  xr[t] = dre[(size_t)b * SS + t * WHN + c] * a;
  xi[t] = dim_[(size_t)b * SS + t * WHN + c] * a;
  __syncthreads();
  const float w = 6.283185307179586f / 128.f;
  float sr = 0.f, si = 0.f;
  for (int k = 0; k < IMGN; ++k) {
    float ang = w * (float)((k * t) & 127);
    float cw = cosf(ang), sw = sinf(ang);
    sr += xr[k] * cw - xi[k] * sw;     // e^{+i ang} (inverse transform)
    si += xr[k] * sw + xi[k] * cw;
  }
  yre[(size_t)b * SS + t * WHN + c] = sr;
  yim[(size_t)b * SS + t * WHN + c] = si;
}

__global__ __launch_bounds__(128) void irfft_row_kernel(
    const float* __restrict__ yre, const float* __restrict__ yim,
    float* __restrict__ img0)
{
  __shared__ float xr[WHN], xi[WHN];
  int br = blockIdx.x;               // b*128 + r
  int b = br >> 7, r = br & 127;
  int t = threadIdx.x;
  if (t < WHN) {
    xr[t] = yre[(size_t)b * SS + r * WHN + t];
    xi[t] = yim[(size_t)b * SS + r * WHN + t];
  }
  __syncthreads();
  const float w = 6.283185307179586f / 128.f;
  float acc = xr[0] + ((t & 1) ? -xr[64] : xr[64]);
  for (int k = 1; k < 64; ++k) {
    float ang = w * (float)((k * t) & 127);
    acc += 2.f * (xr[k] * cosf(ang) - xi[k] * sinf(ang));
  }
  acc *= (1.f / (128.f * 128.f));
  int rr = (r + 64) & 127, mm = (t + 64) & 127;   // jnp.roll fused
  img0[(size_t)b * IMGN * IMGN + rr * IMGN + mm] = acc;
}

// =====================================================================
// Conv refinement: 3x3 conv (1->8)+relu, BN stats, 1x1 conv + BN + residual
// =====================================================================
__global__ void conv1_kernel(const float* __restrict__ img0, const float* __restrict__ w,
                             const float* __restrict__ bias, float* __restrict__ outc)
{
  int idx = blockIdx.x * 256 + threadIdx.x;     // B*8*128*128
  if (idx >= BB * DCN * IMGN * IMGN) return;
  int m  = idx & 127;
  int r  = (idx >> 7) & 127;
  int ch = (idx >> 14) & 7;
  int b  = idx >> 17;
  const float* wp = w + ch * 9;
  float acc = bias[ch];
  #pragma unroll
  for (int dy = -1; dy <= 1; ++dy)
    #pragma unroll
    for (int dx = -1; dx <= 1; ++dx) {
      int rr = r + dy, mm = m + dx;
      if (rr >= 0 && rr < IMGN && mm >= 0 && mm < IMGN)
        acc += wp[(dy + 1) * 3 + (dx + 1)] * img0[(size_t)b * IMGN * IMGN + rr * IMGN + mm];
    }
  outc[idx] = fmaxf(acc, 0.f);
}

__global__ __launch_bounds__(256) void bnstat_kernel(const float* __restrict__ c,
                                                     float* __restrict__ stats)
{
  __shared__ float s1[256], s2[256];
  int ch = blockIdx.x;
  int t = threadIdx.x;
  float a = 0.f, q = 0.f;
  for (int i = t; i < BB * IMGN * IMGN; i += 256) {
    int b = i >> 14;
    int pix = i & 16383;
    float v = c[(((size_t)b * DCN + ch) << 14) + pix];
    a += v; q += v * v;
  }
  s1[t] = a; s2[t] = q;
  __syncthreads();
  for (int off = 128; off > 0; off >>= 1) {
    if (t < off) { s1[t] += s1[t + off]; s2[t] += s2[t + off]; }
    __syncthreads();
  }
  if (t == 0) {
    float m = s1[0] * (1.f / 65536.f);
    stats[ch]     = m;
    stats[8 + ch] = s2[0] * (1.f / 65536.f) - m * m;
  }
}

__global__ void final_kernel(const float* __restrict__ c, const float* __restrict__ stats,
                             const float* __restrict__ g, const float* __restrict__ bb,
                             const float* __restrict__ w2, const float* __restrict__ b2,
                             const float* __restrict__ img0, float* __restrict__ out)
{
  int idx = blockIdx.x * 256 + threadIdx.x;     // B*128*128
  if (idx >= BB * IMGN * IMGN) return;
  int pix = idx & 16383;
  int b   = idx >> 14;
  float acc = b2[0];
  #pragma unroll
  for (int ch = 0; ch < DCN; ++ch) {
    float v = c[(((size_t)b * DCN + ch) << 14) + pix];
    acc += ((v - stats[ch]) * rsqrtf(stats[8 + ch] + 1e-5f) * g[ch] + bb[ch]) * w2[ch];
  }
  out[idx] = acc + img0[idx];
}

// =====================================================================
// Host orchestration
// =====================================================================
extern "C" void kernel_launch(void* const* d_in, const int* in_sizes, int n_in,
                              void* d_out, int out_size, void* d_ws, size_t ws_size,
                              hipStream_t stream)
{
  (void)in_sizes; (void)n_in; (void)out_size; (void)ws_size;
  const float* fbp      = (const float*)d_in[1];
  const float* mag_min  = (const float*)d_in[2];
  const float* mag_max  = (const float*)d_in[3];
  const float* atten    = (const float*)d_in[4];
  const int*   coords   = (const int*)d_in[5];
  const int*   dst      = (const int*)d_in[6];
  const float* emb_w    = (const float*)d_in[8];
  const float* emb_b    = (const float*)d_in[9];
  const float* Wq = (const float*)d_in[10]; const float* bq = (const float*)d_in[11];
  const float* Wk = (const float*)d_in[12]; const float* bk = (const float*)d_in[13];
  const float* Wv = (const float*)d_in[14]; const float* bv = (const float*)d_in[15];
  const float* Wo = (const float*)d_in[16]; const float* bo = (const float*)d_in[17];
  const float* ln1_g = (const float*)d_in[18]; const float* ln1_b = (const float*)d_in[19];
  const float* W1 = (const float*)d_in[20]; const float* b1 = (const float*)d_in[21];
  const float* W2 = (const float*)d_in[22]; const float* b2 = (const float*)d_in[23];
  const float* ln2_g = (const float*)d_in[24]; const float* ln2_b = (const float*)d_in[25];
  const float* lnf_g = (const float*)d_in[26]; const float* lnf_b = (const float*)d_in[27];
  const float* amp_w = (const float*)d_in[28]; const float* amp_b = (const float*)d_in[29];
  const float* ph_w  = (const float*)d_in[30]; const float* ph_b  = (const float*)d_in[31];
  const float* c1w = (const float*)d_in[32]; const float* c1b = (const float*)d_in[33];
  const float* bng = (const float*)d_in[34]; const float* bnb = (const float*)d_in[35];
  const float* c2w = (const float*)d_in[36]; const float* c2b = (const float*)d_in[37];

  float* out = (float*)d_out;            // [0,66560): y_hat; [66560,+65536): img_post
  float* ws  = (float*)d_ws;
  const size_t MSD = (size_t)MM * DD;    // 17,039,360

  float* H    = ws;
  float* Qb   = ws + 1 * MSD;
  float* Kb   = ws + 2 * MSD;
  float* Vb   = ws + 3 * MSD;
  float* Tb   = ws + 4 * MSD;
  float* KV   = ws + 5 * MSD;                    // 32*64*64
  float* KS   = KV + 32 * 64 * 64;               // 32*64
  float* DRE  = KS + 32 * 64;                    // B*S
  float* DIMG = DRE + (size_t)BB * SS;
  float* Y1R  = DIMG + (size_t)BB * SS;
  float* Y1I  = Y1R + (size_t)BB * SS;
  float* IMG0 = Y1I + (size_t)BB * SS;           // B*128*128
  float* CONV = IMG0 + (size_t)BB * IMGN * IMGN; // B*8*128*128
  float* STAT = CONV + (size_t)BB * DCN * IMGN * IMGN; // 16 floats
  // f16 region (16B-aligned: all preceding sizes are multiples of 4 floats)
  _Float16* Hh   = (_Float16*)(STAT + 16);       // M x D
  _Float16* Atth = Hh + MSD;                     // M x D
  _Float16* Midh = Atth + MSD;                   // M x FF
  _Float16* WT   = Midh + (size_t)MM * FFN;      // transposed f16 weights

  const size_t WSTRIDE = 4 * (size_t)DD * DD + 2 * (size_t)DD * FFN; // per layer

  // --- weight convert+transpose (L2-resident f16 weights) ---
  for (int i = 0; i < LL; ++i) {
    _Float16* WqT = WT + i * WSTRIDE;
    _Float16* WkT = WqT + (size_t)DD * DD;
    _Float16* WvT = WkT + (size_t)DD * DD;
    _Float16* WoT = WvT + (size_t)DD * DD;
    _Float16* W1T = WoT + (size_t)DD * DD;
    _Float16* W2T = W1T + (size_t)DD * FFN;
    convT_kernel<<<(DD * DD + 255) / 256, 256, 0, stream>>>(Wq + (size_t)i * DD * DD, WqT, DD, DD);
    convT_kernel<<<(DD * DD + 255) / 256, 256, 0, stream>>>(Wk + (size_t)i * DD * DD, WkT, DD, DD);
    convT_kernel<<<(DD * DD + 255) / 256, 256, 0, stream>>>(Wv + (size_t)i * DD * DD, WvT, DD, DD);
    convT_kernel<<<(DD * DD + 255) / 256, 256, 0, stream>>>(Wo + (size_t)i * DD * DD, WoT, DD, DD);
    convT_kernel<<<(DD * FFN + 255) / 256, 256, 0, stream>>>(W1 + (size_t)i * DD * FFN, W1T, DD, FFN);
    convT_kernel<<<(DD * FFN + 255) / 256, 256, 0, stream>>>(W2 + (size_t)i * DD * FFN, W2T, FFN, DD);
  }

  // --- embedding + PE ---
  embed_kernel<<<(int)((MSD + 255) / 256), 256, 0, stream>>>(fbp, coords, emb_w, emb_b, H, Hh);

  const dim3 gD(DD / GBN, MM / GBM);     // (4, 260)
  const dim3 gF(FFN / GBN, MM / GBM);    // (16, 260)

  for (int i = 0; i < LL; ++i) {
    _Float16* WqT = WT + i * WSTRIDE;
    _Float16* WkT = WqT + (size_t)DD * DD;
    _Float16* WvT = WkT + (size_t)DD * DD;
    _Float16* WoT = WvT + (size_t)DD * DD;
    _Float16* W1T = WoT + (size_t)DD * DD;
    _Float16* W2T = W1T + (size_t)DD * FFN;
    const size_t bo_ = (size_t)i * DD;

    gemm_wmma_kernel<<<gD, 256, 0, stream>>>(Hh, WqT, bq + bo_, Qb, nullptr, MM, DD, DD, 0);
    gemm_wmma_kernel<<<gD, 256, 0, stream>>>(Hh, WkT, bk + bo_, Kb, nullptr, MM, DD, DD, 0);
    gemm_wmma_kernel<<<gD, 256, 0, stream>>>(Hh, WvT, bv + bo_, Vb, nullptr, MM, DD, DD, 0);
    kv_kernel<<<BB * NHH, 256, 0, stream>>>(Kb, Vb, KV, KS);
    att_kernel<<<MM, 256, 0, stream>>>(Qb, KV, KS, Atth);
    gemm_wmma_kernel<<<gD, 256, 0, stream>>>(Atth, WoT, bo + bo_, Qb, nullptr, MM, DD, DD, 0);
    ln_kernel<<<MM, 256, 0, stream>>>(H, Qb, ln1_g + bo_, ln1_b + bo_, H, Hh, 1);
    gemm_wmma_kernel<<<gF, 256, 0, stream>>>(Hh, W1T, b1 + (size_t)i * FFN,
                                             nullptr, Midh, MM, FFN, DD, 1);
    gemm_wmma_kernel<<<gD, 256, 0, stream>>>(Midh, W2T, b2 + bo_, Tb, nullptr, MM, DD, FFN, 0);
    ln_kernel<<<MM, 256, 0, stream>>>(H, Tb, ln2_g + bo_, ln2_b + bo_, H, Hh, 1);
  }

  // --- final LN + heads + DFT scatter ---
  ln_kernel<<<MM, 256, 0, stream>>>(H, nullptr, lnf_g, lnf_b, Tb, nullptr, 0);
  zero_kernel<<<(2 * BB * SS + 255) / 256, 256, 0, stream>>>(DRE, 2 * BB * SS);
  head_kernel<<<MM, 256, 0, stream>>>(Tb, amp_w, amp_b, ph_w, ph_b,
                                      mag_min, mag_max, dst, out, DRE, DIMG);

  // --- inverse FFT ---
  ifft_col_kernel<<<BB * WHN, 128, 0, stream>>>(DRE, DIMG, atten, Y1R, Y1I);
  irfft_row_kernel<<<BB * IMGN, 128, 0, stream>>>(Y1R, Y1I, IMG0);

  // --- conv refinement ---
  conv1_kernel<<<(BB * DCN * IMGN * IMGN + 255) / 256, 256, 0, stream>>>(IMG0, c1w, c1b, CONV);
  bnstat_kernel<<<DCN, 256, 0, stream>>>(CONV, STAT);
  final_kernel<<<(BB * IMGN * IMGN + 255) / 256, 256, 0, stream>>>(
      CONV, STAT, bng, bnb, c2w, c2b, IMG0, out + (size_t)BB * SS * 2);
}
